// SelfAttention_67542655697730
// MI455X (gfx1250) — compile-verified, matrix operand order checked by reading, not run on hardware
//
#include <hip/hip_runtime.h>
#include <hip/hip_bf16.h>
#include <stdint.h>

typedef __attribute__((ext_vector_type(16))) __bf16 v16bf;
typedef __attribute__((ext_vector_type(8)))  __bf16 v8bf;
typedef __attribute__((ext_vector_type(8)))  float  v8f;
typedef __attribute__((ext_vector_type(4)))  float  v4f;
typedef __attribute__((ext_vector_type(4)))  unsigned int u32x4;
typedef __attribute__((ext_vector_type(8)))  int  i32x8;
typedef __attribute__((ext_vector_type(4)))  int  i32x4;

#define C_DIM 512
#define D_DIM 64
#define N_PIX 4096
#define BATCH 4
#define BM 32          // queries per workgroup
#define BN 64          // keys per block iteration

#if __has_builtin(__builtin_amdgcn_tensor_load_to_lds)
#define USE_TDM 1
#else
#define USE_TDM 0
#endif

static __device__ inline v16bf cat8(v8bf a, v8bf b) {
    return __builtin_shufflevector(a, b, 0,1,2,3,4,5,6,7,8,9,10,11,12,13,14,15);
}
static __device__ inline v16bf cvt16(v4f a, v4f b, v4f c, v4f d) {
    v8f lo = {a[0],a[1],a[2],a[3],b[0],b[1],b[2],b[3]};
    v8f hi = {c[0],c[1],c[2],c[3],d[0],d[1],d[2],d[3]};
    return cat8(__builtin_convertvector(lo, v8bf), __builtin_convertvector(hi, v8bf));
}
static __device__ inline v8f wmma_bf16(v16bf a, v16bf b, v8f c) {
    return __builtin_amdgcn_wmma_f32_16x16x32_bf16(false, a, false, b, (short)0, c, false, false);
}

#if USE_TDM
// ---- Tensor Data Mover: build D# per CDNA5 ISA ----
// This toolchain's builtin is the 6-arg form:
//   (u32x4 group0, i32x8 group1, i32x4 group2, i32x4 group3, i32x8 extra, i32 cpol)
// We only use a 2-D tile (groups 2/3 unused -> zero).
static __device__ inline void tdm_load_2d(uint32_t lds_off, const void* gsrc,
                                          uint32_t tile_d0, uint32_t tile_d1,
                                          uint32_t tens_d0, uint32_t tens_d1,
                                          uint64_t stride0_elems)
{
    uint64_t ga = (uint64_t)(uintptr_t)gsrc;
    u32x4 g0;
    g0[0] = 1u;                                     // count=1, is_restore=0, gather=0
    g0[1] = lds_off;                                // lds_addr (bytes)
    g0[2] = (uint32_t)ga;                           // global_addr[31:0]
    g0[3] = (uint32_t)((ga >> 32) & 0x01FFFFFFu) | (2u << 30);  // addr[56:32] | type=2
    i32x8 g1;
    g1[0] = (int)(1u << 16);                        // data_size=1 (2 bytes)
    g1[1] = (int)((tens_d0 & 0xFFFFu) << 16);       // tensor_dim0[15:0] -> bits 63:48
    g1[2] = (int)((tens_d0 >> 16) | ((tens_d1 & 0xFFFFu) << 16));
    g1[3] = (int)((tens_d1 >> 16) | (tile_d0 << 16));
    g1[4] = (int)(tile_d1 & 0xFFFFu);               // tile_dim1; tile_dim2=0
    g1[5] = (int)(uint32_t)(stride0_elems & 0xFFFFFFFFu);
    g1[6] = (int)(uint32_t)((stride0_elems >> 32) & 0xFFFFu);   // dim1_stride unused (2D)
    g1[7] = 0;
    i32x4 z4 = {0,0,0,0};
    i32x8 z8 = {0,0,0,0,0,0,0,0};
    __builtin_amdgcn_tensor_load_to_lds(g0, g1, z4, z4, z8, 0);
}
static __device__ inline void tdm_wait() {
#if __has_builtin(__builtin_amdgcn_s_wait_tensorcnt)
    __builtin_amdgcn_s_wait_tensorcnt(0);
#else
    asm volatile("s_wait_tensorcnt 0x0" ::: "memory");
#endif
}
#endif

// Fallback cooperative copies (also used if TDM builtin is unavailable)
static __device__ inline void coop_copy_k(__bf16* dst, const __bf16* src, int tid) {
    for (int i = tid; i < (BN * D_DIM) / 8; i += 256)           // contiguous 8 KB
        ((v8bf*)dst)[i] = ((const v8bf*)src)[i];
}
static __device__ inline void coop_copy_v(__bf16* dst, const __bf16* src, int tid) {
    for (int i = tid; i < C_DIM * (BN / 8); i += 256) {
        int row = i >> 3, ch = i & 7;
        *(v8bf*)(dst + row * BN + ch * 8) =
            *(const v8bf*)(src + (size_t)row * N_PIX + ch * 8);
    }
}

// ---------------------------------------------------------------------------
// Prepass: x [b][c][n] f32  ->  xbT [b][n][c] bf16 (transpose + convert once).
// ---------------------------------------------------------------------------
__global__ __launch_bounds__(256)
void transpose_bf16_kernel(const float* __restrict__ x, __bf16* __restrict__ xbT)
{
    __shared__ float tile[32][33];
    const int b  = blockIdx.z;
    const int c0 = blockIdx.y * 32;
    const int n0 = blockIdx.x * 32;
    const int tx = threadIdx.x & 31;
    const int g  = threadIdx.x >> 5;      // 8 groups of 4 rows
    #pragma unroll
    for (int i = 0; i < 4; ++i) {
        int c = g * 4 + i;
        tile[c][tx] = x[((size_t)b * C_DIM + c0 + c) * N_PIX + n0 + tx];
    }
    __syncthreads();
    #pragma unroll
    for (int i = 0; i < 4; ++i) {
        int n = g * 4 + i;
        xbT[((size_t)b * N_PIX + n0 + n) * C_DIM + c0 + tx] = (__bf16)tile[tx][n];
    }
}

// ---------------------------------------------------------------------------
// q/k projection: outT[n][d] = sum_c xbT[n][c] * W[d][c] + bias[d].
// A = xbT rows (contiguous c chunks), B = W cols (contiguous c chunks, f32->bf16).
// Output qT/kT [b][n][64] bf16 with lane-coalesced 2B stores.
// ---------------------------------------------------------------------------
__global__ __launch_bounds__(256)
void proj_qk_kernel(const __bf16* __restrict__ xbT, const float* __restrict__ W,
                    const float* __restrict__ bias, __bf16* __restrict__ outT)
{
    const int lane = threadIdx.x & 31, wv = threadIdx.x >> 5;
    const int l15 = lane & 15, hi = lane >> 4, koff = hi * 8;
    const int n0 = blockIdx.x * 128 + wv * 16;   // pixel rows
    const int d0 = blockIdx.y * 16;              // output-dim cols
    const int b  = blockIdx.z;

    const __bf16* xrow = xbT + ((size_t)b * N_PIX + n0 + l15) * C_DIM;
    const float*  wrow = W + (size_t)(d0 + l15) * C_DIM;

    v8f acc = {};
    for (int k0 = 0; k0 < C_DIM; k0 += 32) {
        v16bf a = cat8(*(const v8bf*)(xrow + k0 + koff),
                       *(const v8bf*)(xrow + k0 + 16 + koff));
        v16bf bw = cvt16(*(const v4f*)(wrow + k0 + koff),
                         *(const v4f*)(wrow + k0 + koff + 4),
                         *(const v4f*)(wrow + k0 + 16 + koff),
                         *(const v4f*)(wrow + k0 + 16 + koff + 4));
        acc = wmma_bf16(a, bw, acc);
    }
    float bb = bias[d0 + l15];
    #pragma unroll
    for (int r = 0; r < 8; ++r) {
        int n = n0 + r + (hi << 3);
        outT[((size_t)b * N_PIX + n) * D_DIM + d0 + l15] = (__bf16)(acc[r] + bb);
    }
}

// ---------------------------------------------------------------------------
// v projection: out[d][n] = sum_c W[d][c] * xbT[n][c] + bias[d].
// A = W rows (contiguous c), B = xbT cols (contiguous c). Coalesced [c][n] store.
// ---------------------------------------------------------------------------
__global__ __launch_bounds__(256)
void proj_v_kernel(const __bf16* __restrict__ xbT, const float* __restrict__ W,
                   const float* __restrict__ bias, __bf16* __restrict__ out)
{
    const int lane = threadIdx.x & 31, wv = threadIdx.x >> 5;
    const int l15 = lane & 15, hi = lane >> 4, koff = hi * 8;
    const int n0 = blockIdx.x * 128 + wv * 16;   // pixel cols
    const int d0 = blockIdx.y * 16;              // channel rows
    const int b  = blockIdx.z;

    const float*  wrow = W + (size_t)(d0 + l15) * C_DIM;
    const __bf16* xrow = xbT + ((size_t)b * N_PIX + n0 + l15) * C_DIM;

    v8f acc = {};
    for (int k0 = 0; k0 < C_DIM; k0 += 32) {
        v16bf aw = cvt16(*(const v4f*)(wrow + k0 + koff),
                         *(const v4f*)(wrow + k0 + koff + 4),
                         *(const v4f*)(wrow + k0 + 16 + koff),
                         *(const v4f*)(wrow + k0 + 16 + koff + 4));
        v16bf bx = cat8(*(const v8bf*)(xrow + k0 + koff),
                        *(const v8bf*)(xrow + k0 + 16 + koff));
        acc = wmma_bf16(aw, bx, acc);
    }
    #pragma unroll
    for (int r = 0; r < 8; ++r) {
        int d = d0 + r + (hi << 3);
        out[((size_t)b * C_DIM + d) * (size_t)N_PIX + n0 + l15] = (__bf16)(acc[r] + bias[d]);
    }
}

// ---------------------------------------------------------------------------
// Flash attention: 32 queries/WG, 64-key blocks, TDM double-buffered k/v tiles.
// Wave (g,t): g = query half (16 rows), t = key sub-tile / 128-channel slice.
// Per iter/wave: 2 WMMA (scores) + 16 WMMA (PV).
// ---------------------------------------------------------------------------
__global__ __launch_bounds__(256)
void attn_flash_kernel(const __bf16* __restrict__ qT,  // [B][N][64]
                       const __bf16* __restrict__ kT,  // [B][N][64]
                       const __bf16* __restrict__ vC,  // [B][512][N]
                       const float*  __restrict__ x,   // [B][512][N]
                       float* __restrict__ out)        // [B][512][N]
{
    constexpr int OFF_K  = 0;                     // 2 * 64*64*2   = 16384
    constexpr int OFF_V  = 16384;                 // 2 * 512*64*2  = 131072
    constexpr int OFF_P  = OFF_V + 131072;        // 32*64*2       = 4096
    constexpr int OFF_RD = OFF_P + 4096;          // 2 * 4*32*4    = 1024
    constexpr int OFF_ST = OFF_RD + 1024;         // 4 * 32*4      = 512
    __shared__ __align__(16) unsigned char smem[OFF_ST + 512];   // ~150 KB

    __bf16 (*kb)[BN][D_DIM] = (__bf16 (*)[BN][D_DIM])(smem + OFF_K);
    __bf16 (*vb)[C_DIM][BN] = (__bf16 (*)[C_DIM][BN])(smem + OFF_V);
    __bf16 (*Pb)[BN]        = (__bf16 (*)[BN])(smem + OFF_P);
    float* red_max = (float*)(smem + OFF_RD);     // [4][32]
    float* red_sum = red_max + 4 * 32;
    float* m_run   = (float*)(smem + OFF_ST);
    float* l_run   = m_run + 32;
    float* m_cur   = l_run + 32;
    float* scale_s = m_cur + 32;

    const int tid = threadIdx.x;
    const int lane = tid & 31, wv = tid >> 5;
    const int l15 = lane & 15, hi = lane >> 4, koff = hi * 8;
    const int g  = wv >> 2;             // query half
    const int t4 = wv & 3;              // key sub-tile / channel slice
    const int m0 = blockIdx.x * BM;
    const int b  = blockIdx.y;
    const int c0 = t4 * 128;

    const __bf16* ksrc = kT + (size_t)b * N_PIX * D_DIM;
    const __bf16* vsrc = vC + (size_t)b * C_DIM * N_PIX;

    // ---- prologue: stage block 0 into buffer 0 (TDM), init stats, load Q ----
#if USE_TDM
    if (wv == 0) {
        tdm_load_2d((uint32_t)(uintptr_t)&kb[0][0][0], ksrc, D_DIM, BN, D_DIM, N_PIX, D_DIM);
        tdm_load_2d((uint32_t)(uintptr_t)&vb[0][0][0], vsrc, BN, C_DIM, N_PIX, C_DIM, N_PIX);
    }
#else
    coop_copy_k(&kb[0][0][0], ksrc, tid);
    coop_copy_v(&vb[0][0][0], vsrc, tid);
#endif
    if (tid < 32) { m_run[tid] = -3.0e38f; l_run[tid] = 0.0f; }

    const __bf16* qrow = qT + ((size_t)b * N_PIX + m0 + g * 16 + l15) * D_DIM;
    v16bf aq0 = cat8(*(const v8bf*)(qrow + koff),      *(const v8bf*)(qrow + 16 + koff));
    v16bf aq1 = cat8(*(const v8bf*)(qrow + 32 + koff), *(const v8bf*)(qrow + 48 + koff));

    v8f acc_o[8] = {};
#if USE_TDM
    if (wv == 0) tdm_wait();
#endif
    __syncthreads();

    for (int j0 = 0; j0 < N_PIX; j0 += BN) {
        const int buf  = (j0 / BN) & 1;
        const int nbuf = buf ^ 1;

        // async prefetch of next k/v block into the other buffer
        if (j0 + BN < N_PIX) {
#if USE_TDM
            if (wv == 0) {
                tdm_load_2d((uint32_t)(uintptr_t)&kb[nbuf][0][0],
                            ksrc + (size_t)(j0 + BN) * D_DIM, D_DIM, BN, D_DIM, N_PIX, D_DIM);
                tdm_load_2d((uint32_t)(uintptr_t)&vb[nbuf][0][0],
                            vsrc + (j0 + BN), BN, C_DIM, N_PIX, C_DIM, N_PIX);
            }
#else
            coop_copy_k(&kb[nbuf][0][0], ksrc + (size_t)(j0 + BN) * D_DIM, tid);
            coop_copy_v(&vb[nbuf][0][0], vsrc + (j0 + BN), tid);
#endif
        }

        // ---------------- scores S = Q.K^T (16x16 per wave) -----------------
        const __bf16* krow = &kb[buf][t4 * 16 + l15][0];
        v16bf bk0 = cat8(*(const v8bf*)(krow + koff),      *(const v8bf*)(krow + 16 + koff));
        v16bf bk1 = cat8(*(const v8bf*)(krow + 32 + koff), *(const v8bf*)(krow + 48 + koff));
        v8f s = {};
        s = wmma_bf16(aq0, bk0, s);
        s = wmma_bf16(aq1, bk1, s);

        // per-row tile max (reduce across 16 lanes per half-wave)
        float rmx[8];
        #pragma unroll
        for (int r = 0; r < 8; ++r) rmx[r] = s[r];
        #pragma unroll
        for (int off = 1; off < 16; off <<= 1)
            #pragma unroll
            for (int r = 0; r < 8; ++r)
                rmx[r] = fmaxf(rmx[r], __shfl_xor(rmx[r], off, 16));
        if (l15 == 0)
            #pragma unroll
            for (int r = 0; r < 8; ++r)
                red_max[t4 * 32 + g * 16 + r + (hi << 3)] = rmx[r];
        __syncthreads();

        if (tid < 32) {                  // one thread per query row (32 rows)
            int row = tid;
            float mb = red_max[row];
            #pragma unroll
            for (int t = 1; t < 4; ++t) mb = fmaxf(mb, red_max[t * 32 + row]);
            float mo = m_run[row];
            float mn = fmaxf(mo, mb);
            m_cur[row]   = mn;
            scale_s[row] = __expf(mo - mn);
            m_run[row]   = mn;
        }
        __syncthreads();

        // ---------------- P = exp(S - m), LDS tile, row sums ----------------
        float rsum[8];
        #pragma unroll
        for (int r = 0; r < 8; ++r) {
            int row = g * 16 + r + (hi << 3);
            float p = __expf(s[r] - m_cur[row]);
            Pb[row][t4 * 16 + l15] = (__bf16)p;
            rsum[r] = p;
        }
        #pragma unroll
        for (int off = 1; off < 16; off <<= 1)
            #pragma unroll
            for (int r = 0; r < 8; ++r)
                rsum[r] += __shfl_xor(rsum[r], off, 16);
        if (l15 == 0)
            #pragma unroll
            for (int r = 0; r < 8; ++r)
                red_sum[t4 * 32 + g * 16 + r + (hi << 3)] = rsum[r];
        __syncthreads();

        if (tid < 32) {
            int row = tid;
            float sb = 0.f;
            #pragma unroll
            for (int t = 0; t < 4; ++t) sb += red_sum[t * 32 + row];
            l_run[row] = l_run[row] * scale_s[row] + sb;
        }

        // ---------------- PV: O[16q x 128c slice] += P.V^T ------------------
        float sc[8];
        #pragma unroll
        for (int r = 0; r < 8; ++r) sc[r] = scale_s[g * 16 + r + (hi << 3)];
        #pragma unroll
        for (int tl = 0; tl < 8; ++tl)
            #pragma unroll
            for (int r = 0; r < 8; ++r) acc_o[tl][r] *= sc[r];

        #pragma unroll
        for (int kk = 0; kk < 2; ++kk) {
            const __bf16* prow = &Pb[g * 16 + l15][kk * 32];
            v16bf ap = cat8(*(const v8bf*)(prow + koff), *(const v8bf*)(prow + 16 + koff));
            #pragma unroll
            for (int tl = 0; tl < 8; ++tl) {
                const __bf16* vrow = &vb[buf][c0 + tl * 16 + l15][kk * 32];
                v16bf bvv = cat8(*(const v8bf*)(vrow + koff), *(const v8bf*)(vrow + 16 + koff));
                acc_o[tl] = wmma_bf16(ap, bvv, acc_o[tl]);
            }
        }
#if USE_TDM
        if (wv == 0) tdm_wait();
#endif
        __syncthreads();
    }

    // ---- epilogue: normalize, transpose tiles via LDS so stores/residual
    //      reads are coalesced along m (64B runs), add residual, store fp32 ---
    float* stage = (float*)(smem + OFF_V) + wv * (16 * 17);
    float linv[8];
    #pragma unroll
    for (int r = 0; r < 8; ++r) linv[r] = 1.0f / l_run[g * 16 + r + (hi << 3)];

    #pragma unroll
    for (int tl = 0; tl < 8; ++tl) {
        #pragma unroll
        for (int r = 0; r < 8; ++r)
            stage[(r + (hi << 3)) * 17 + l15] = acc_o[tl][r] * linv[r];
        __syncthreads();
        const int m = m0 + g * 16 + l15;
        #pragma unroll
        for (int rr = 0; rr < 8; ++rr) {
            int c = c0 + tl * 16 + rr + (hi << 3);
            size_t idx = ((size_t)b * C_DIM + c) * (size_t)N_PIX + m;
            out[idx] = stage[l15 * 17 + rr + (hi << 3)] + x[idx];
        }
        __syncthreads();
    }
}

// ---------------------------------------------------------------------------
extern "C" void kernel_launch(void* const* d_in, const int* in_sizes, int n_in,
                              void* d_out, int out_size, void* d_ws, size_t ws_size,
                              hipStream_t stream)
{
    const float* x  = (const float*)d_in[0];
    const float* Wq = (const float*)d_in[1];
    const float* bq = (const float*)d_in[2];
    const float* Wk = (const float*)d_in[3];
    const float* bk = (const float*)d_in[4];
    const float* Wv = (const float*)d_in[5];
    const float* bv = (const float*)d_in[6];
    float* out = (float*)d_out;

    __bf16* xbT = (__bf16*)d_ws;                                  // 16 MB
    __bf16* qT  = xbT + (size_t)BATCH * N_PIX * C_DIM;            //  2 MB
    __bf16* kT  = qT  + (size_t)BATCH * N_PIX * D_DIM;            //  2 MB
    __bf16* vB  = kT  + (size_t)BATCH * N_PIX * D_DIM;            // 16 MB

    // Prepass: transpose + bf16 conversion of x (once, shared by all GEMMs)
    transpose_bf16_kernel<<<dim3(N_PIX / 32, C_DIM / 32, BATCH), 256, 0, stream>>>(x, xbT);

    // Projections (1x1 convs as WMMA GEMMs, all-vector fragment loads)
    proj_qk_kernel<<<dim3(N_PIX / 128, D_DIM / 16, BATCH), 256, 0, stream>>>(xbT, Wq, bq, qT);
    proj_qk_kernel<<<dim3(N_PIX / 128, D_DIM / 16, BATCH), 256, 0, stream>>>(xbT, Wk, bk, kT);
    proj_v_kernel <<<dim3(N_PIX / 128, C_DIM / 16, BATCH), 256, 0, stream>>>(xbT, Wv, bv, vB);

    // Flash attention + residual (TDM double-buffered k/v tiles)
    attn_flash_kernel<<<dim3(N_PIX / BM, BATCH), 256, 0, stream>>>(qT, kT, vB, x, out);
}
